// TPALinearQKV_25632364822890
// MI455X (gfx1250) — matrix operand order repaired and checked by reading, not compiled, not run on hardware
//
#include <hip/hip_runtime.h>

// Problem dims (fixed by the reference)
#define D_MODEL 2048
#define M_TOK   8192            // S*B = 4096*2 tokens
#define N_COLS  1408            // 96 + 16 + 16 + 768 + 256 + 256
#define KT_N    64              // 2048 / 32 k-tiles
#define OUT_PER_TOK 4096        // 8 groups * (256 q + 128 k + 128 v)

typedef __bf16 bf16_t;
typedef __attribute__((ext_vector_type(16))) __bf16 v16bf;
typedef __attribute__((ext_vector_type(8)))  float  v8f;

// ---------------------------------------------------------------------------
// Pack X (fp32 row-major [8192,2048]) into bf16 WMMA A-fragment tiles.
// Tile (mt,kt) = 16(M) x 32(K); within a tile: 32 lanes x 16 halves contiguous.
// ISA 16-bit A 16x32 layout: lane l<16 holds row M=l, K={0..7,16..23};
// lane l+16 holds row M=l, K={8..15,24..31}.  k_local = (h&7)+((h>>3)<<4)+((lane>>4)<<3)
// ---------------------------------------------------------------------------
__global__ __launch_bounds__(256) void pack_a_kernel(const float* __restrict__ X,
                                                     bf16_t* __restrict__ Ap) {
    int idx  = blockIdx.x * 256 + threadIdx.x;       // one packed element
    int h    = idx & 15;
    int lane = (idx >> 4) & 31;
    int tile = idx >> 9;                             // mt*KT_N + kt
    int kt   = tile & (KT_N - 1);
    int mt   = tile >> 6;
    int m = (mt << 4) + (lane & 15);
    int k = (kt << 5) + (h & 7) + ((h >> 3) << 4) + ((lane >> 4) << 3);
    Ap[idx] = (bf16_t)X[(size_t)m * D_MODEL + k];
}

// ---------------------------------------------------------------------------
// Pack the six fp32 weights (each row-major [2048, out]) into bf16 WMMA
// B-fragment tiles.  Tile (nt,kt) = 32(K) x 16(N); lane n<16 holds column
// N=n, K={0..7,16..23} across the 16 halves; lane n+16 holds K={8..15,24..31}.
// Column-concatenated order: a_q | a_k | a_v | b_q | b_k | b_v.
// ---------------------------------------------------------------------------
__global__ __launch_bounds__(256) void pack_b_kernel(const float* __restrict__ Waq,
                                                     const float* __restrict__ Wak,
                                                     const float* __restrict__ Wav,
                                                     const float* __restrict__ Wbq,
                                                     const float* __restrict__ Wbk,
                                                     const float* __restrict__ Wbv,
                                                     bf16_t* __restrict__ Bp) {
    int idx  = blockIdx.x * 256 + threadIdx.x;
    int h    = idx & 15;
    int lane = (idx >> 4) & 31;
    int tile = idx >> 9;                             // nt*KT_N + kt
    int kt   = tile & (KT_N - 1);
    int nt   = tile >> 6;
    int n = (nt << 4) + (lane & 15);
    int k = (kt << 5) + (h & 7) + ((h >> 3) << 4) + ((lane >> 4) << 3);
    float v;
    if      (n <   96) v = Waq[(size_t)k *  96 +  n        ];
    else if (n <  112) v = Wak[(size_t)k *  16 + (n -   96)];
    else if (n <  128) v = Wav[(size_t)k *  16 + (n -  112)];
    else if (n <  896) v = Wbq[(size_t)k * 768 + (n -  128)];
    else if (n < 1152) v = Wbk[(size_t)k * 256 + (n -  896)];
    else               v = Wbv[(size_t)k * 256 + (n - 1152)];
    Bp[idx] = (bf16_t)v;
}

// ---------------------------------------------------------------------------
// C[8192,1408] = X @ Wcat, bf16 in / f32 accumulate.
// Block = 4 waves (128 thr); block tile 128x128; each wave a 64x64 tile
// (4x4 accumulators of v_wmma_f32_16x16x32_bf16).  64 k-steps of K=32.
// Fragments are contiguous 32 B/lane loads from the packed buffers.
// ---------------------------------------------------------------------------
__global__ __launch_bounds__(128) void tpa_gemm_kernel(const bf16_t* __restrict__ Ap,
                                                       const bf16_t* __restrict__ Bp,
                                                       float* __restrict__ C) {
    const int lane = threadIdx.x & 31;
    const int wave = threadIdx.x >> 5;
    const int wm   = wave >> 1;
    const int wn   = wave & 1;
    const int mt0  = blockIdx.y * 8 + wm * 4;        // base 16-row tile
    const int nt0  = blockIdx.x * 8 + wn * 4;        // base 16-col tile

    v8f acc[4][4];
    const v8f vzero = {0.f, 0.f, 0.f, 0.f, 0.f, 0.f, 0.f, 0.f};
#pragma unroll
    for (int i = 0; i < 4; ++i)
#pragma unroll
        for (int j = 0; j < 4; ++j) acc[i][j] = vzero;

    const bf16_t* aB = Ap + (size_t)mt0 * KT_N * 512 + lane * 16;
    const bf16_t* bB = Bp + (size_t)nt0 * KT_N * 512 + lane * 16;

    for (int kt = 0; kt < KT_N; ++kt) {
        if (kt + 8 < KT_N) {                         // stream-ahead hints
            __builtin_prefetch(aB + ((size_t)0 * KT_N + kt + 8) * 512, 0, 1);
            __builtin_prefetch(bB + ((size_t)0 * KT_N + kt + 8) * 512, 0, 1);
        }
        v16bf a[4], b[4];
#pragma unroll
        for (int i = 0; i < 4; ++i)
            a[i] = *(const v16bf*)(aB + ((size_t)i * KT_N + kt) * 512);
#pragma unroll
        for (int j = 0; j < 4; ++j)
            b[j] = *(const v16bf*)(bB + ((size_t)j * KT_N + kt) * 512);
#pragma unroll
        for (int i = 0; i < 4; ++i)
#pragma unroll
            for (int j = 0; j < 4; ++j)
                acc[i][j] = __builtin_amdgcn_wmma_f32_16x16x32_bf16(
                    /*neg_a=*/false, a[i], /*neg_b=*/false, b[j],
                    /*c_mod=*/(short)0, acc[i][j],
                    /*reuse_a=*/false, /*reuse_b=*/false);
    }

    // C/D 16x16 f32 layout: VGPR r -> M = r (lanes 0-15) or r+8 (lanes 16-31),
    // N = lane & 15.
    const int rowOff = (lane >> 4) << 3;
    const int col    = lane & 15;
#pragma unroll
    for (int i = 0; i < 4; ++i) {
#pragma unroll
        for (int r = 0; r < 8; ++r) {
            size_t m = (size_t)(mt0 + i) * 16 + r + rowOff;
            float* crow = C + m * N_COLS + (size_t)nt0 * 16 + col;
#pragma unroll
            for (int j = 0; j < 4; ++j) crow[j * 16] = acc[i][j][r];
        }
    }
}

// ---------------------------------------------------------------------------
// Per-token low-rank reconstruction + interleave.
// Row layout in C: [a_q 96 | a_k 16 | a_v 16 | b_q 768 | b_k 256 | b_v 256].
// Output per token (4096): for g in 0..7: [q[2g](128) q[2g+1](128) k[g](128) v[g](128)]
// ---------------------------------------------------------------------------
__global__ __launch_bounds__(128) void tpa_recon_kernel(const float* __restrict__ C,
                                                        float* __restrict__ out) {
    __shared__ float sA[128];    // a_q[16][6] | a_k[8][2] | a_v[8][2]
    __shared__ float sB[1280];   // b_q[6][128] | b_k[2][128] | b_v[2][128]
    const size_t t   = blockIdx.x;
    const float* row = C + t * N_COLS;
    const int tid    = threadIdx.x;

    sA[tid] = row[tid];
    for (int i = tid; i < 1280; i += 128) sB[i] = row[128 + i];
    __syncthreads();

    float* o = out + t * OUT_PER_TOK;
    for (int it = 0; it < OUT_PER_TOK / 128; ++it) {
        int c = it * 128 + tid;
        int g = c >> 9;
        int w = c & 511;
        float val;
        if (w < 256) {                               // q heads 2g, 2g+1
            int h = g * 2 + (w >> 7);
            int d = w & 127;
            float s = 0.f;
#pragma unroll
            for (int r = 0; r < 6; ++r) s += sA[h * 6 + r] * sB[r * 128 + d];
            val = s * (1.0f / 6.0f);
        } else if (w < 384) {                        // k[g]
            int d = w - 256;
            float s = sA[96 + g * 2 + 0] * sB[768 + 0 * 128 + d]
                    + sA[96 + g * 2 + 1] * sB[768 + 1 * 128 + d];
            val = s * 0.5f;
        } else {                                     // v[g]
            int d = w - 384;
            float s = sA[112 + g * 2 + 0] * sB[1024 + 0 * 128 + d]
                    + sA[112 + g * 2 + 1] * sB[1024 + 1 * 128 + d];
            val = s * 0.5f;
        }
        o[c] = val;
    }
}

// ---------------------------------------------------------------------------
extern "C" void kernel_launch(void* const* d_in, const int* in_sizes, int n_in,
                              void* d_out, int out_size, void* d_ws, size_t ws_size,
                              hipStream_t stream) {
    const float* X   = (const float*)d_in[0];
    const float* Waq = (const float*)d_in[1];
    const float* Wak = (const float*)d_in[2];
    const float* Wav = (const float*)d_in[3];
    const float* Wbq = (const float*)d_in[4];
    const float* Wbk = (const float*)d_in[5];
    const float* Wbv = (const float*)d_in[6];
    float* out = (float*)d_out;

    // Workspace layout (bytes):
    //   Apack bf16 : 8192*2048*2  = 33,554,432
    //   Bpack bf16 : 2048*1408*2  =  5,767,168
    //   C     f32  : 8192*1408*4  = 46,137,344   (total ~85.5 MB)
    char*   ws = (char*)d_ws;
    bf16_t* Ap = (bf16_t*)(ws);
    bf16_t* Bp = (bf16_t*)(ws + (size_t)33554432);
    float*  C  = (float*) (ws + (size_t)33554432 + 5767168);

    pack_a_kernel<<<(M_TOK * D_MODEL) / 256, 256, 0, stream>>>(X, Ap);
    pack_b_kernel<<<(D_MODEL * N_COLS) / 256, 256, 0, stream>>>(Waq, Wak, Wav, Wbq, Wbk, Wbv, Bp);
    tpa_gemm_kernel<<<dim3(N_COLS / 128, M_TOK / 128), 128, 0, stream>>>(Ap, Bp, C);
    tpa_recon_kernel<<<M_TOK, 128, 0, stream>>>(C, out);
}
